// DGraphCast_58007828299999
// MI455X (gfx1250) — matrix operand order repaired and checked by reading, not compiled, not run on hardware
//
#include <hip/hip_runtime.h>
#include <cstdint>
#include <cstddef>

#define GRID_N 65536
#define MESH_N 40962
#define M2M_E 327680
#define G2M_E 262144
#define M2G_E 196608
#define GRID_IN 474
#define OUT_DIM 227
#define HID 128

typedef __bf16 v16bf __attribute__((ext_vector_type(16)));
typedef float  v8f   __attribute__((ext_vector_type(8)));
typedef float  f32x4 __attribute__((ext_vector_type(4)));
typedef unsigned int u32x4 __attribute__((ext_vector_type(4)));
typedef unsigned int u32x2 __attribute__((ext_vector_type(2)));

union FragAB { u32x4 q[2]; unsigned int u[8]; v16bf v; };

// f32 -> bf16 round-to-nearest-even (bit twiddle; no dependence on cast codegen)
__device__ __forceinline__ unsigned int f2bf(float x) {
  unsigned int u = __builtin_bit_cast(unsigned int, x);
  unsigned int r = (u + 0x7FFFu + ((u >> 16) & 1u)) >> 16;
  return r & 0xFFFFu;
}

// ---------------------------------------------------------------------------
// Weight prep: f32 [din x dout] -> bf16 WMMA B-fragment layout.
// Per (kt, nt) tile: 32 lanes x 8 dwords. Lane L: column N = nt*16 + (L%16);
// dword j holds K = kt*32 + (L/16)*16 + 2j and +1  (contiguous K order; matches fused_mlp A reads).
// ---------------------------------------------------------------------------
__global__ __launch_bounds__(256) void prep_w(const float* __restrict__ w,
                                              int din, int dout, int ntiles, int kts,
                                              unsigned int* __restrict__ frag) {
  int i = blockIdx.x * 256 + threadIdx.x;
  int total = kts * ntiles * 256;
  if (i >= total) return;
  int kt  = i / (ntiles * 256);
  int rem = i - kt * ntiles * 256;
  int nt  = rem >> 8;
  int lr  = rem & 255;
  int lane = lr >> 3, j = lr & 7;
  int n  = nt * 16 + (lane & 15);
  int kb = kt * 32 + ((lane >> 4) << 4) + (j << 1);
  float lo = 0.0f, hi = 0.0f;
  if (n < dout) {
    if (kb < din)     lo = w[(long)kb * dout + n];
    if (kb + 1 < din) hi = w[(long)(kb + 1) * dout + n];
  }
  frag[i] = f2bf(lo) | (f2bf(hi) << 16);
}

// ---------------------------------------------------------------------------
// Fused MLP: out = [LN]( SiLU( cat(src0[i0],src1[i1],src2[i2]) @ W1 + b1 ) @ W2 + b2 ) [+ resid]
// Block: 256 threads (8 waves), 32 rows (2 M-tiles -> each B fragment feeds 2 WMMAs).
// Hidden = 128 (8 N-tiles, one per wave).
// LDS: sBuf aliases input bf16 tile (phase 1) and f32 output tile (phase 2).
// ---------------------------------------------------------------------------
struct FArgs {
  const float* src0; const int* idx0; int w0;
  const float* src1; const int* idx1; int w1;
  const float* src2; const int* idx2; int w2;
  int R;
  const unsigned int* W1f; int Kpad;          // pad32(w0+w1+w2)
  const float* b1;
  const unsigned int* W2f; int dout; int nt2; // nt2 = ceil(dout/16)
  const float* b2;
  const float* gamma; const float* beta;      // null => no LayerNorm
  const float* resid;                         // null => no residual
  float* out;
};

__global__ __launch_bounds__(256) void fused_mlp(FArgs a) {
  // 30720 B shared between: bf16 input tile [32 x Kpad<=480]  (phase 1)
  //                         f32 output tile [32 x dpad<=240]  (phase 2)
  __shared__ __align__(16) unsigned char sBuf[32 * 480 * 2];
  __shared__ __align__(16) unsigned short sH[32 * HID];   // bf16 hidden tile
  __shared__ int sIdx[96];
  unsigned short* sX = (unsigned short*)sBuf;
  float* sO = (float*)sBuf;

  const int t = threadIdx.x;
  const int row0 = (int)blockIdx.x * 32;
  const int cw = a.w0 + a.w1 + a.w2;

  if (t < 96) {
    int p = t >> 5, r = t & 31;
    int gr = row0 + r;
    const int* ip = (p == 0) ? a.idx0 : (p == 1) ? a.idx1 : a.idx2;
    int rr = gr;
    if (ip && gr < a.R) rr = ip[gr];
    sIdx[p * 32 + r] = rr;
  }
  __syncthreads();

  // ---- Gather + convert concat rows into LDS (zero pad K and OOB rows) ----
  const bool vec4 = (((a.w0 | a.w1 | a.w2) & 3) == 0);
  if (vec4) {
    // All segment widths multiple of 4: float4 loads, bf16x4 (8B) LDS stores.
    const int kq = a.Kpad >> 2;
    const int ng = 32 * kq;
    for (int i = t; i < ng; i += 256) {
      int r = i / kq, c = (i - r * kq) << 2;
      int gr = row0 + r;
      f32x4 v = {0.0f, 0.0f, 0.0f, 0.0f};
      if (gr < a.R && c < cw) {
        const float* sp;
        if (c < a.w0)             sp = a.src0 + (long)sIdx[r]      * a.w0 + c;
        else if (c < a.w0 + a.w1) sp = a.src1 + (long)sIdx[32 + r] * a.w1 + (c - a.w0);
        else                      sp = a.src2 + (long)sIdx[64 + r] * a.w2 + (c - a.w0 - a.w1);
        v = *(const f32x4*)sp;
      }
      u32x2 d;
      d.x = f2bf(v.x) | (f2bf(v.y) << 16);
      d.y = f2bf(v.z) | (f2bf(v.w) << 16);
      *(u32x2*)&sX[r * a.Kpad + c] = d;
    }
  } else {
    const int nx = 32 * a.Kpad;
    for (int i = t; i < nx; i += 256) {
      int r = i / a.Kpad, c = i - r * a.Kpad;
      int gr = row0 + r;
      float v = 0.0f;
      if (gr < a.R && c < cw) {
        if (c < a.w0)             v = a.src0[(long)sIdx[r]      * a.w0 + c];
        else if (c < a.w0 + a.w1) v = a.src1[(long)sIdx[32 + r] * a.w1 + (c - a.w0)];
        else                      v = a.src2[(long)sIdx[64 + r] * a.w2 + (c - a.w0 - a.w1)];
      }
      sX[i] = (unsigned short)f2bf(v);
    }
  }
  __syncthreads();

  const int wv = t >> 5, lane = t & 31;
  const int colN = lane & 15;
  const int rA   = lane & 15;            // A row within M-tile
  const int kOff = (lane >> 4) << 4;     // 0 or 16 : A K-offset (contiguous K, matches B)
  const int hrow = (lane >> 4) << 3;     // C/D rows: VGPR i -> M = hrow + i

  // ---- GEMM1: X(32 x Kpad) @ W1 -> two 16x16 hidden tiles per wave ----
  v8f acc0, acc1;
  for (int i = 0; i < 8; ++i) { acc0[i] = 0.0f; acc1[i] = 0.0f; }
  const int kts = a.Kpad >> 5;
  for (int kt = 0; kt < kts; ++kt) {
    FragAB A0, A1, B;
    const u32x4* wp = (const u32x4*)(a.W1f + ((((kt * 8 + wv) * 32) + lane) << 3));
    B.q[0] = wp[0]; B.q[1] = wp[1];
    const unsigned short* x0 = &sX[rA * a.Kpad + kt * 32 + kOff];
    const unsigned short* x1 = &sX[(rA + 16) * a.Kpad + kt * 32 + kOff];
    A0.q[0] = *(const u32x4*)(x0); A0.q[1] = *(const u32x4*)(x0 + 8);
    A1.q[0] = *(const u32x4*)(x1); A1.q[1] = *(const u32x4*)(x1 + 8);
    acc0 = __builtin_amdgcn_wmma_f32_16x16x32_bf16(false, A0.v, false, B.v,
                                                   (short)0, acc0, false, false);
    acc1 = __builtin_amdgcn_wmma_f32_16x16x32_bf16(false, A1.v, false, B.v,
                                                   (short)0, acc1, false, false);
  }
  {
    float bb = a.b1[wv * 16 + colN];
    for (int i = 0; i < 8; ++i) {
      float x0 = acc0[i] + bb;
      float x1 = acc1[i] + bb;
      x0 = x0 / (1.0f + __expf(-x0));     // SiLU
      x1 = x1 / (1.0f + __expf(-x1));
      sH[(hrow + i)      * HID + wv * 16 + colN] = (unsigned short)f2bf(x0);
      sH[(hrow + i + 16) * HID + wv * 16 + colN] = (unsigned short)f2bf(x1);
    }
  }
  __syncthreads();   // also separates sX reads from aliased sO writes

  // ---- GEMM2: H(32x128) @ W2 -> out tiles (wave-strided over N tiles) ----
  const int dpad = a.nt2 * 16;
  for (int nt = wv; nt < a.nt2; nt += 8) {
    v8f c0, c1;
    for (int i = 0; i < 8; ++i) { c0[i] = 0.0f; c1[i] = 0.0f; }
    for (int kt = 0; kt < 4; ++kt) {
      FragAB A0, A1, B;
      const u32x4* wp = (const u32x4*)(a.W2f + ((((kt * a.nt2 + nt) * 32) + lane) << 3));
      B.q[0] = wp[0]; B.q[1] = wp[1];
      const unsigned short* x0 = &sH[rA * HID + kt * 32 + kOff];
      const unsigned short* x1 = &sH[(rA + 16) * HID + kt * 32 + kOff];
      A0.q[0] = *(const u32x4*)(x0); A0.q[1] = *(const u32x4*)(x0 + 8);
      A1.q[0] = *(const u32x4*)(x1); A1.q[1] = *(const u32x4*)(x1 + 8);
      c0 = __builtin_amdgcn_wmma_f32_16x16x32_bf16(false, A0.v, false, B.v,
                                                   (short)0, c0, false, false);
      c1 = __builtin_amdgcn_wmma_f32_16x16x32_bf16(false, A1.v, false, B.v,
                                                   (short)0, c1, false, false);
    }
    int col = nt * 16 + colN;
    float b2v = (col < a.dout) ? a.b2[col] : 0.0f;
    for (int i = 0; i < 8; ++i) {
      sO[(hrow + i)      * dpad + col] = c0[i] + b2v;
      sO[(hrow + i + 16) * dpad + col] = c1[i] + b2v;
    }
  }
  __syncthreads();

  // ---- Epilogue: LayerNorm (+resid) or plain (+resid) ----
  if (a.gamma) {
    for (int half = 0; half < 32; half += 16) {
      int r = (t >> 4) + half, l = t & 15;
      float s = 0.0f, s2 = 0.0f;
      for (int c = l; c < a.dout; c += 16) { float x = sO[r * dpad + c]; s += x; s2 += x * x; }
      for (int mr = 8; mr >= 1; mr >>= 1) {
        s  += __shfl_xor(s,  mr, 16);
        s2 += __shfl_xor(s2, mr, 16);
      }
      float inv = 1.0f / (float)a.dout;
      float mean = s * inv;
      float var  = s2 * inv - mean * mean;
      float rstd = rsqrtf(var + 1e-5f);
      int gr = row0 + r;
      if (gr < a.R) {
        for (int c = l; c < a.dout; c += 16) {
          float x = (sO[r * dpad + c] - mean) * rstd * a.gamma[c] + a.beta[c];
          if (a.resid) x += a.resid[(long)gr * a.dout + c];
          a.out[(long)gr * a.dout + c] = x;
        }
      }
    }
  } else {
    for (int i = t; i < 32 * a.dout; i += 256) {
      int r = i / a.dout, c = i - r * a.dout;
      int gr = row0 + r;
      if (gr < a.R) {
        float x = sO[r * dpad + c];
        if (a.resid) x += a.resid[(long)gr * a.dout + c];
        a.out[(long)gr * a.dout + c] = x;
      }
    }
  }
}

// ---------------------------------------------------------------------------
__global__ __launch_bounds__(256) void zero_f32(float* p, long n) {
  long i = (long)blockIdx.x * 256 + threadIdx.x;
  if (i < n) p[i] = 0.0f;
}

__global__ __launch_bounds__(256) void add_f32(const float* __restrict__ x,
                                               const float* __restrict__ y,
                                               float* __restrict__ o, long n) {
  long i = (long)blockIdx.x * 256 + threadIdx.x;
  if (i < n) o[i] = x[i] + y[i];
}

// segment_sum: agg[idx[e]] += e_feat[e], one thread per (edge, col)
__global__ __launch_bounds__(256) void scatter_add(const float* __restrict__ e,
                                                   const int* __restrict__ idx,
                                                   float* __restrict__ agg, long E) {
  long i = (long)blockIdx.x * 256 + threadIdx.x;
  if (i >= E * HID) return;
  int ed = (int)(i >> 7), c = (int)(i & 127);
  __hip_atomic_fetch_add(&agg[(long)idx[ed] * HID + c], e[i],
                         __ATOMIC_RELAXED, __HIP_MEMORY_SCOPE_AGENT);
}

// ---------------------------------------------------------------------------
// Host orchestration
// ---------------------------------------------------------------------------
// params flattened in JAX tree order (alphabetical): per-MLP {b,b1,b2,g,w1,w2},
// MLPs: dec_edge, dec_node, emb_g2m, emb_grid, emb_m2g, emb_m2m, emb_mesh,
//       enc_edge, enc_grid, enc_node, final, proc_edge[0..2], proc_node[0..2]
static const int MLP_DIN[17]  = {384,256,4,474,4,4,3,384,128,256,128,384,384,384,256,256,256};
static const int MLP_DOUT[17] = {128,128,128,128,128,128,128,128,128,128,227,128,128,128,128,128,128};
enum { ML_DEC_EDGE = 0, ML_DEC_NODE, ML_EMB_G2M, ML_EMB_GRID, ML_EMB_M2G, ML_EMB_M2M,
       ML_EMB_MESH, ML_ENC_EDGE, ML_ENC_GRID, ML_ENC_NODE, ML_FINAL,
       ML_PROC_E0, ML_PROC_E1, ML_PROC_E2, ML_PROC_N0, ML_PROC_N1, ML_PROC_N2 };

extern "C" void kernel_launch(void* const* d_in, const int* in_sizes, int n_in,
                              void* d_out, int out_size, void* d_ws, size_t ws_size,
                              hipStream_t stream) {
  (void)in_sizes; (void)n_in; (void)out_size; (void)ws_size;

  const float* grid_in = (const float*)d_in[0];
  const float* mesh_in = (const float*)d_in[1];
  const float* m2m_f   = (const float*)d_in[2];
  const float* g2m_f   = (const float*)d_in[3];
  const float* m2g_f   = (const float*)d_in[4];
  const int* m2m_src = (const int*)d_in[5];
  const int* m2m_dst = (const int*)d_in[6];
  const int* g2m_src = (const int*)d_in[7];
  const int* g2m_dst = (const int*)d_in[8];
  const int* m2g_src = (const int*)d_in[9];
  const int* m2g_dst = (const int*)d_in[10];

  struct MP { const float *b, *b1, *b2, *g, *w1, *w2;
              unsigned int *w1f, *w2f; int din, dout, kts1, nt2; };
  MP m[17];

  size_t off = 0;
  char* ws = (char*)d_ws;
  auto alloc = [&](size_t bytes) -> void* {
    void* p = ws + off;
    off += (bytes + 255) & ~(size_t)255;
    return p;
  };

  for (int i = 0; i < 17; ++i) {
    int base = 11 + 6 * i;
    m[i].b  = (const float*)d_in[base + 0];
    m[i].b1 = (const float*)d_in[base + 1];
    m[i].b2 = (const float*)d_in[base + 2];
    m[i].g  = (const float*)d_in[base + 3];
    m[i].w1 = (const float*)d_in[base + 4];
    m[i].w2 = (const float*)d_in[base + 5];
    m[i].din  = MLP_DIN[i];
    m[i].dout = MLP_DOUT[i];
    m[i].kts1 = (m[i].din + 31) / 32;
    m[i].nt2  = (m[i].dout + 15) / 16;
    m[i].w1f = (unsigned int*)alloc((size_t)m[i].kts1 * 8 * 256 * 4);
    m[i].w2f = (unsigned int*)alloc((size_t)4 * m[i].nt2 * 256 * 4);
  }

  float* eg     = (float*)alloc((size_t)GRID_N * HID * 4);
  float* em     = (float*)alloc((size_t)MESH_N * HID * 4);
  float* e_m2m  = (float*)alloc((size_t)M2M_E * HID * 4);
  float* e_g2m  = (float*)alloc((size_t)G2M_E * HID * 4);
  float* e_m2g  = (float*)alloc((size_t)M2G_E * HID * 4);
  float* agg_m  = (float*)alloc((size_t)MESH_N * HID * 4);
  float* tmp_m  = (float*)alloc((size_t)MESH_N * HID * 4);
  float* mesh_n = (float*)alloc((size_t)MESH_N * HID * 4);
  float* grid_n = (float*)alloc((size_t)GRID_N * HID * 4);
  float* agg_g  = (float*)alloc((size_t)GRID_N * HID * 4);
  float* tmp_g  = (float*)alloc((size_t)GRID_N * HID * 4);
  float* grid_o = (float*)alloc((size_t)GRID_N * HID * 4);

  // Weight swizzle to bf16 WMMA fragments (every call; deterministic)
  for (int i = 0; i < 17; ++i) {
    int t1 = m[i].kts1 * 8 * 256;
    prep_w<<<(t1 + 255) / 256, 256, 0, stream>>>(m[i].w1, m[i].din, HID, 8, m[i].kts1, m[i].w1f);
    int t2 = 4 * m[i].nt2 * 256;
    prep_w<<<(t2 + 255) / 256, 256, 0, stream>>>(m[i].w2, HID, m[i].dout, m[i].nt2, 4, m[i].w2f);
  }

  auto run = [&](const MP& p,
                 const float* s0, const int* i0, int w0,
                 const float* s1, const int* i1, int w1,
                 const float* s2, const int* i2, int w2,
                 int R, const float* resid, bool ln, float* out) {
    FArgs a;
    a.src0 = s0; a.idx0 = i0; a.w0 = w0;
    a.src1 = s1; a.idx1 = i1; a.w1 = w1;
    a.src2 = s2; a.idx2 = i2; a.w2 = w2;
    a.R = R; a.W1f = p.w1f; a.Kpad = p.kts1 * 32; a.b1 = p.b1;
    a.W2f = p.w2f; a.dout = p.dout; a.nt2 = p.nt2; a.b2 = p.b2;
    a.gamma = ln ? p.g : nullptr;
    a.beta  = ln ? p.b : nullptr;
    a.resid = resid; a.out = out;
    fused_mlp<<<(R + 31) / 32, 256, 0, stream>>>(a);
  };

  // ---- embedders ----
  run(m[ML_EMB_GRID], grid_in, nullptr, GRID_IN, nullptr, nullptr, 0, nullptr, nullptr, 0,
      GRID_N, nullptr, true, eg);
  run(m[ML_EMB_MESH], mesh_in, nullptr, 3, nullptr, nullptr, 0, nullptr, nullptr, 0,
      MESH_N, nullptr, true, em);
  run(m[ML_EMB_M2M], m2m_f, nullptr, 4, nullptr, nullptr, 0, nullptr, nullptr, 0,
      M2M_E, nullptr, true, e_m2m);
  run(m[ML_EMB_G2M], g2m_f, nullptr, 4, nullptr, nullptr, 0, nullptr, nullptr, 0,
      G2M_E, nullptr, true, e_g2m);
  run(m[ML_EMB_M2G], m2g_f, nullptr, 4, nullptr, nullptr, 0, nullptr, nullptr, 0,
      M2G_E, nullptr, true, e_m2g);

  const long nm = (long)MESH_N * HID;
  const long ng = (long)GRID_N * HID;

  // ---- encoder (grid -> mesh) ----
  run(m[ML_ENC_EDGE], e_g2m, nullptr, HID, eg, g2m_src, HID, em, g2m_dst, HID,
      G2M_E, e_g2m, true, e_g2m);
  zero_f32<<<(int)((nm + 255) / 256), 256, 0, stream>>>(agg_m, nm);
  {
    long se = (long)G2M_E * HID;
    scatter_add<<<(int)((se + 255) / 256), 256, 0, stream>>>(e_g2m, g2m_dst, agg_m, (long)G2M_E);
  }
  run(m[ML_ENC_NODE], agg_m, nullptr, HID, em, nullptr, HID, nullptr, nullptr, 0,
      MESH_N, em, true, tmp_m);
  add_f32<<<(int)((nm + 255) / 256), 256, 0, stream>>>(em, tmp_m, mesh_n, nm);
  run(m[ML_ENC_GRID], eg, nullptr, HID, nullptr, nullptr, 0, nullptr, nullptr, 0,
      GRID_N, eg, true, grid_n);

  // ---- processor (mesh -> mesh), 3 layers ----
  for (int L = 0; L < 3; ++L) {
    run(m[ML_PROC_E0 + L], e_m2m, nullptr, HID, mesh_n, m2m_src, HID, mesh_n, m2m_dst, HID,
        M2M_E, e_m2m, true, e_m2m);
    zero_f32<<<(int)((nm + 255) / 256), 256, 0, stream>>>(agg_m, nm);
    long sm = (long)M2M_E * HID;
    scatter_add<<<(int)((sm + 255) / 256), 256, 0, stream>>>(e_m2m, m2m_src, agg_m, (long)M2M_E);
    run(m[ML_PROC_N0 + L], agg_m, nullptr, HID, mesh_n, nullptr, HID, nullptr, nullptr, 0,
        MESH_N, mesh_n, true, mesh_n);
  }

  // ---- decoder (mesh -> grid) ----
  run(m[ML_DEC_EDGE], e_m2g, nullptr, HID, mesh_n, m2g_src, HID, grid_n, m2g_dst, HID,
      M2G_E, e_m2g, true, e_m2g);
  zero_f32<<<(int)((ng + 255) / 256), 256, 0, stream>>>(agg_g, ng);
  {
    long sg = (long)M2G_E * HID;
    scatter_add<<<(int)((sg + 255) / 256), 256, 0, stream>>>(e_m2g, m2g_dst, agg_g, (long)M2G_E);
  }
  run(m[ML_DEC_NODE], agg_g, nullptr, HID, grid_n, nullptr, HID, nullptr, nullptr, 0,
      GRID_N, grid_n, true, tmp_g);
  add_f32<<<(int)((ng + 255) / 256), 256, 0, stream>>>(grid_n, tmp_g, grid_o, ng);

  // ---- final head (no LayerNorm, no residual) ----
  run(m[ML_FINAL], grid_o, nullptr, HID, nullptr, nullptr, 0, nullptr, nullptr, 0,
      GRID_N, nullptr, false, (float*)d_out);
}
